// MixFormerBlock_81200651698760
// MI455X (gfx1250) — matrix operand order repaired and checked by reading, not compiled, not run on hardware
//
#include <hip/hip_runtime.h>
#include <math.h>

typedef _Float16 half_t;
typedef __attribute__((ext_vector_type(16))) _Float16 v16h;
typedef __attribute__((ext_vector_type(8)))  _Float16 v8h;
typedef __attribute__((ext_vector_type(8)))  float    v8f;
typedef __attribute__((ext_vector_type(4)))  unsigned v4u;
typedef __attribute__((ext_vector_type(8)))  int      v8i;
typedef __attribute__((ext_vector_type(4)))  int      v4i;

#define WMMA_F16(a, b, c) \
  __builtin_amdgcn_wmma_f32_16x16x32_f16(false, (a), false, (b), (short)0, (c), false, false)

#define N_TOK 980
#define CH    768
#define NHEAD 12
#define HDIM  64
#define QKVC  2304
#define HIDC  3072
#define MROWS 15680   /* B * N_TOK */
#define TLEN  196     /* t_h * t_w = 14*14 */
#define SSTR  992     /* padded key extent: ceil(980/32)*32 */

// ---------------------------------------------------------------------------
// Weight convert + transpose: fp32 [K, Nc] -> f16 [Nc, K]
// ---------------------------------------------------------------------------
__global__ void k_cvt_transpose(const float* __restrict__ w, half_t* __restrict__ wt,
                                int K, int Nc) {
  long idx = (long)blockIdx.x * blockDim.x + threadIdx.x;
  long total = (long)K * Nc;
  if (idx >= total) return;
  int k = (int)(idx / Nc);
  int n = (int)(idx % Nc);
  wt[(long)n * K + k] = (half_t)w[idx];
}

// ---------------------------------------------------------------------------
// LayerNorm: fp32 [rows, 768] -> f16 [rows, 768]; one block (256 thr) per row
// ---------------------------------------------------------------------------
__global__ __launch_bounds__(256) void k_layernorm(const float* __restrict__ x,
                                                   const float* __restrict__ w,
                                                   const float* __restrict__ b,
                                                   half_t* __restrict__ y) {
  __shared__ float red[256];
  int row = blockIdx.x;
  int t = threadIdx.x;
  const float* xr = x + (long)row * CH;
  float v0 = xr[t], v1 = xr[t + 256], v2 = xr[t + 512];
  red[t] = v0 + v1 + v2;
  __syncthreads();
  for (int off = 128; off > 0; off >>= 1) {
    if (t < off) red[t] += red[t + off];
    __syncthreads();
  }
  float mean = red[0] * (1.0f / CH);
  __syncthreads();
  float d0 = v0 - mean, d1 = v1 - mean, d2 = v2 - mean;
  red[t] = d0 * d0 + d1 * d1 + d2 * d2;
  __syncthreads();
  for (int off = 128; off > 0; off >>= 1) {
    if (t < off) red[t] += red[t + off];
    __syncthreads();
  }
  float rstd = rsqrtf(red[0] * (1.0f / CH) + 1e-5f);
  half_t* yr = y + (long)row * CH;
  yr[t]       = (half_t)(d0 * rstd * w[t]       + b[t]);
  yr[t + 256] = (half_t)(d1 * rstd * w[t + 256] + b[t + 256]);
  yr[t + 512] = (half_t)(d2 * rstd * w[t + 512] + b[t + 512]);
}

// ---------------------------------------------------------------------------
// Wave-tiled WMMA GEMM: C[M,Nc] = A[M,K](f16) @ WT[Nc,K]^T(f16) + bias
//   mode 0: store f16 | mode 1: exact GELU, store f16 | mode 2: +fp32 residual, store fp32
// Each wave computes a 16(M) x 64(N) tile; block = 8 waves.
// ---------------------------------------------------------------------------
__global__ __launch_bounds__(256) void k_gemm(const half_t* __restrict__ A,
                                              const half_t* __restrict__ WT,
                                              const float* __restrict__ bias,
                                              const float* __restrict__ resid,
                                              half_t* __restrict__ outh,
                                              float* __restrict__ outf,
                                              int Mtiles, int K, int Nc, int mode) {
  int wave = (int)((blockIdx.x * blockDim.x + threadIdx.x) >> 5);
  int lane = threadIdx.x & 31;
  int ntiles = Nc >> 6;
  if (wave >= Mtiles * ntiles) return;
  int mt = wave / ntiles;
  int nt = wave % ntiles;
  int n16 = lane & 15;
  int hh  = lane >> 4;

  const half_t* arow = A + (long)(mt * 16 + n16) * K;
  long wcol0 = (long)(nt * 64 + n16) * K;

  v8f acc0 = {0,0,0,0,0,0,0,0};
  v8f acc1 = {0,0,0,0,0,0,0,0};
  v8f acc2 = {0,0,0,0,0,0,0,0};
  v8f acc3 = {0,0,0,0,0,0,0,0};
  union { v16h v; v8h h[2]; } af;

  for (int k0 = 0; k0 < K; k0 += 32) {
    af.h[0] = *(const v8h*)(arow + k0 + hh * 16);
    af.h[1] = *(const v8h*)(arow + k0 + hh * 16 + 8);
    const half_t* wp = WT + wcol0 + k0 + hh * 16;
    v16h b0 = *(const v16h*)(wp);
    v16h b1 = *(const v16h*)(wp + (long)16 * K);
    v16h b2 = *(const v16h*)(wp + (long)32 * K);
    v16h b3 = *(const v16h*)(wp + (long)48 * K);
    acc0 = WMMA_F16(af.v, b0, acc0);
    acc1 = WMMA_F16(af.v, b1, acc1);
    acc2 = WMMA_F16(af.v, b2, acc2);
    acc3 = WMMA_F16(af.v, b3, acc3);
  }

  v8f accs[4] = {acc0, acc1, acc2, acc3};
  for (int s = 0; s < 4; s++) {
    int col = nt * 64 + s * 16 + n16;
    float bb = bias[col];
    for (int j = 0; j < 8; j++) {
      int row = mt * 16 + j + 8 * hh;
      float v = accs[s][j] + bb;
      if (mode == 1) v = 0.5f * v * (1.0f + erff(v * 0.70710678118f));
      if (mode == 2) {
        outf[(long)row * Nc + col] = v + resid[(long)row * Nc + col];
      } else {
        outh[(long)row * Nc + col] = (half_t)v;
      }
    }
  }
}

// ---------------------------------------------------------------------------
// TDM helper: async-load a 2D f16 tile [rows=32, cols=64] (row stride 2304
// elements) from global to LDS. Rows beyond `rows_avail` are zero-filled by
// the TDM OOB rule. D# built per CDNA5 ISA 8.3/8.4; 6-arg toolchain builtin.
// ---------------------------------------------------------------------------
__device__ __forceinline__ void tdm_load_v_tile(const half_t* gsrc, unsigned lds_byte_addr,
                                                unsigned rows_avail) {
  unsigned long long ga = (unsigned long long)(size_t)gsrc;  // byte address
  v4u g0;
  g0[0] = 1u;                                        // count=1, user descriptor
  g0[1] = lds_byte_addr;                             // lds_addr [63:32]
  g0[2] = (unsigned)(ga & 0xFFFFFFFFu);              // global_addr lo
  g0[3] = (unsigned)((ga >> 32) & 0x01FFFFFFu)       // global_addr [56:32]
        | (2u << 30);                                // type = 2 ("image")
  v8i g1;
  g1[0] = (int)(1u << 16);                           // data_size=1 -> 2 bytes
  g1[1] = (int)(64u << 16);                          // tensor_dim0 = 64  [79:48]
  g1[2] = (int)((rows_avail & 0xFFFFu) << 16);       // tensor_dim1 lo    [111:80]
  g1[3] = (int)(((rows_avail >> 16) & 0xFFFFu) | (64u << 16)); // tdim1 hi | tile_dim0=64
  g1[4] = 32;                                        // tile_dim1 = 32 rows
  g1[5] = QKVC;                                      // tensor_dim0_stride lo (elements)
  g1[6] = 0;                                         // stride hi | tensor_dim1_stride lo
  g1[7] = 0;
  v4i z4 = {0, 0, 0, 0};
  v8i z8 = {0, 0, 0, 0, 0, 0, 0, 0};
  __builtin_amdgcn_tensor_load_to_lds(g0, g1, z4, z4, z8, 0);
}

// ---------------------------------------------------------------------------
// Attention: one wave per (b, h, 16-query tile).
// qkv f16 layout [B*N, 2304]: col = which*768 + h*64 + d.
// out f16 [B*N, 768].
// ---------------------------------------------------------------------------
__global__ __launch_bounds__(32) void k_attn(const half_t* __restrict__ qkv,
                                             half_t* __restrict__ out,
                                             int q_start, int q_end, int kv_len) {
  __shared__ half_t S[16 * SSTR];       // 31744 B: scores then exp(scores)
  __shared__ half_t Vt[2][32 * HDIM];   // 2 x 4096 B: TDM-staged V tiles
  __shared__ float  rcp[16];
  int b = blockIdx.z, h = blockIdx.y, qt = blockIdx.x;
  int lane = threadIdx.x;
  int n16 = lane & 15;
  int hh  = lane >> 4;
  int q0 = q_start + qt * 16;
  const float scale = 0.125f;  // 64^-0.5
  int nk32 = (kv_len + 31) / 32;
  int padded = nk32 * 32;

  // Q fragments (d split 0..31, 32..63), loaded once
  const half_t* qrow = qkv + (long)(b * N_TOK + q0 + n16) * QKVC + h * HDIM;
  union { v16h v; v8h h2[2]; } aq0, aq1;
  aq0.h2[0] = *(const v8h*)(qrow + 0  + hh * 16);
  aq0.h2[1] = *(const v8h*)(qrow + 8  + hh * 16);
  aq1.h2[0] = *(const v8h*)(qrow + 32 + hh * 16);
  aq1.h2[1] = *(const v8h*)(qrow + 40 + hh * 16);

  // ----- S = scale * Q @ K^T, masked, into LDS (f16) -----
  const half_t* kbase = qkv + CH + h * HDIM;
  for (int kt = 0; kt < padded / 16; kt++) {
    int key = kt * 16 + n16;
    int kc = key < kv_len ? key : kv_len - 1;
    const half_t* krow = kbase + (long)(b * N_TOK + kc) * QKVC + hh * 16;
    v16h b0 = *(const v16h*)(krow);
    v16h b1 = *(const v16h*)(krow + 32);
    v8f c = {0,0,0,0,0,0,0,0};
    c = WMMA_F16(aq0.v, b0, c);
    c = WMMA_F16(aq1.v, b1, c);
    for (int j = 0; j < 8; j++) {
      int m = j + 8 * hh;
      S[m * SSTR + key] = (key < kv_len) ? (half_t)(c[j] * scale) : (half_t)(-60000.0f);
    }
  }

  // ----- row softmax numerator (lanes 0..15 own a row; wave-lockstep) -----
  if (lane < 16) {
    half_t* row = S + lane * SSTR;
    float mx = -60000.0f;
    for (int j = 0; j < padded; j++) mx = fmaxf(mx, (float)row[j]);
    float sum = 0.0f;
    for (int j = 0; j < padded; j++) {
      float e = __expf((float)row[j] - mx);
      row[j] = (half_t)e;         // unnormalized probability
      sum += e;
    }
    rcp[lane] = 1.0f / sum;       // folded into output store
  }

  // ----- O = P @ V with TDM-staged, double-buffered V tiles -----
  const half_t* vbase = qkv + 2 * CH + h * HDIM;
  v8f oacc[4];
  for (int s = 0; s < 4; s++) oacc[s] = (v8f){0,0,0,0,0,0,0,0};

  tdm_load_v_tile(vbase + (long)(b * N_TOK) * QKVC,
                  (unsigned)(size_t)(void*)&Vt[0][0], (unsigned)kv_len);

  for (int kt = 0; kt < nk32; kt++) {
    int kb = kt * 32;
    if (kt + 1 < nk32) {
      int kb2 = kb + 32;
      tdm_load_v_tile(vbase + (long)(b * N_TOK + kb2) * QKVC,
                      (unsigned)(size_t)(void*)&Vt[(kt + 1) & 1][0],
                      (unsigned)(kv_len - kb2));
      __builtin_amdgcn_s_wait_tensorcnt(1);
    } else {
      __builtin_amdgcn_s_wait_tensorcnt(0);
    }

    // A fragment: two direct 16B LDS vector loads of unnormalized P
    const half_t* srow = S + n16 * SSTR + kb;
    union { v16h v; v8h h2[2]; } pa;
    pa.h2[0] = *(const v8h*)(srow + hh * 16);
    pa.h2[1] = *(const v8h*)(srow + hh * 16 + 8);

    const half_t* vt = &Vt[kt & 1][0];
    for (int s = 0; s < 4; s++) {
      int dcol = s * 16 + n16;
      v16h vb;
      for (int v = 0; v < 8; v++) {
        int r0 = hh * 16 + 2 * v;
        vb[2 * v]     = vt[(r0)     * HDIM + dcol];
        vb[2 * v + 1] = vt[(r0 + 1) * HDIM + dcol];
      }
      oacc[s] = WMMA_F16(pa.v, vb, oacc[s]);
    }
  }

  for (int s = 0; s < 4; s++) {
    for (int j = 0; j < 8; j++) {
      int qr = q0 + j + 8 * hh;
      if (qr < q_end) {
        float r = rcp[j + 8 * hh];
        out[(long)(b * N_TOK + qr) * CH + h * HDIM + s * 16 + n16] =
            (half_t)(oacc[s][j] * r);
      }
    }
  }
}

// ---------------------------------------------------------------------------
// Host orchestration
// ---------------------------------------------------------------------------
extern "C" void kernel_launch(void* const* d_in, const int* in_sizes, int n_in,
                              void* d_out, int out_size, void* d_ws, size_t ws_size,
                              hipStream_t stream) {
  const float* x      = (const float*)d_in[0];
  const float* ln1_w  = (const float*)d_in[5];
  const float* ln1_b  = (const float*)d_in[6];
  const float* qkv_w  = (const float*)d_in[7];
  const float* qkv_b  = (const float*)d_in[8];
  const float* proj_w = (const float*)d_in[9];
  const float* proj_b = (const float*)d_in[10];
  const float* ln2_w  = (const float*)d_in[11];
  const float* ln2_b  = (const float*)d_in[12];
  const float* fc1_w  = (const float*)d_in[13];
  const float* fc1_b  = (const float*)d_in[14];
  const float* fc2_w  = (const float*)d_in[15];
  const float* fc2_b  = (const float*)d_in[16];
  float* out = (float*)d_out;

  char* ws = (char*)d_ws;
  size_t off = 0;
  auto alloc = [&](size_t bytes) {
    size_t o = off;
    off += (bytes + 255) & ~(size_t)255;
    return o;
  };
  size_t o_wtqkv = alloc((size_t)CH * QKVC * 2);
  size_t o_wtprj = alloc((size_t)CH * CH * 2);
  size_t o_wtfc1 = alloc((size_t)CH * HIDC * 2);
  size_t o_wtfc2 = alloc((size_t)HIDC * CH * 2);
  size_t o_bufA  = alloc((size_t)MROWS * CH * 2);    // ln1 -> attn out -> ln2
  size_t o_big   = alloc((size_t)MROWS * HIDC * 2);  // qkv f16, then fc1 hidden
  size_t o_x1    = alloc((size_t)MROWS * CH * 4);    // fp32 residual after attn

  half_t* wtqkv = (half_t*)(ws + o_wtqkv);
  half_t* wtprj = (half_t*)(ws + o_wtprj);
  half_t* wtfc1 = (half_t*)(ws + o_wtfc1);
  half_t* wtfc2 = (half_t*)(ws + o_wtfc2);
  half_t* bufA  = (half_t*)(ws + o_bufA);
  half_t* big   = (half_t*)(ws + o_big);
  float*  x1    = (float*)(ws + o_x1);

  // 1) weight convert + transpose
  k_cvt_transpose<<<(CH * QKVC + 255) / 256, 256, 0, stream>>>(qkv_w, wtqkv, CH, QKVC);
  k_cvt_transpose<<<(CH * CH + 255) / 256, 256, 0, stream>>>(proj_w, wtprj, CH, CH);
  k_cvt_transpose<<<(CH * HIDC + 255) / 256, 256, 0, stream>>>(fc1_w, wtfc1, CH, HIDC);
  k_cvt_transpose<<<(HIDC * CH + 255) / 256, 256, 0, stream>>>(fc2_w, wtfc2, HIDC, CH);

  // 2) LN1 -> bufA (f16)
  k_layernorm<<<MROWS, 256, 0, stream>>>(x, ln1_w, ln1_b, bufA);

  // 3) qkv = ln1 @ qkv_w + b  (f16 out into big)
  {
    int waves = (MROWS / 16) * (QKVC / 64);
    k_gemm<<<(waves + 7) / 8, 256, 0, stream>>>(bufA, wtqkv, qkv_b, nullptr,
                                                big, nullptr, MROWS / 16, CH, QKVC, 0);
  }

  // 4) attention -> bufA (f16)
  k_attn<<<dim3((TLEN + 15) / 16, NHEAD, 16), 32, 0, stream>>>(big, bufA, 0, TLEN, TLEN);
  k_attn<<<dim3((N_TOK - TLEN) / 16, NHEAD, 16), 32, 0, stream>>>(big, bufA, TLEN, N_TOK, N_TOK);

  // 5) x1 = attn @ proj_w + b + x   (fp32)
  {
    int waves = (MROWS / 16) * (CH / 64);
    k_gemm<<<(waves + 7) / 8, 256, 0, stream>>>(bufA, wtprj, proj_b, x,
                                                nullptr, x1, MROWS / 16, CH, CH, 2);
  }

  // 6) LN2 -> bufA (f16)
  k_layernorm<<<MROWS, 256, 0, stream>>>(x1, ln2_w, ln2_b, bufA);

  // 7) h = gelu(ln2 @ fc1_w + b)  (f16 out into big)
  {
    int waves = (MROWS / 16) * (HIDC / 64);
    k_gemm<<<(waves + 7) / 8, 256, 0, stream>>>(bufA, wtfc1, fc1_b, nullptr,
                                                big, nullptr, MROWS / 16, CH, HIDC, 1);
  }

  // 8) out = h @ fc2_w + b + x1   (fp32)
  {
    int waves = (MROWS / 16) * (CH / 64);
    k_gemm<<<(waves + 7) / 8, 256, 0, stream>>>(big, wtfc2, fc2_b, x1,
                                                nullptr, out, MROWS / 16, HIDC, CH, 2);
  }
}
